// CharGPT_52063593562741
// MI455X (gfx1250) — matrix operand order repaired and checked by reading, not compiled, not run on hardware
//
#include <hip/hip_runtime.h>

// ---------------- model dims (fixed by reference) ----------------
#define Vv  256
#define Cc  1024
#define Hh  16
#define HSc 64
#define Ll  6
#define Tt  1024
#define Bb  8
#define Mm  (Bb * Tt)   // 8192 rows

typedef __attribute__((ext_vector_type(16))) _Float16 v16h;
typedef __attribute__((ext_vector_type(8)))  _Float16 v8h;
typedef __attribute__((ext_vector_type(8)))  float    v8f;
typedef __attribute__((ext_vector_type(4)))  unsigned int v4u;
typedef __attribute__((ext_vector_type(4)))  int      v4i;
typedef __attribute__((ext_vector_type(8)))  int      v8i;

// ---------------- WMMA helpers ----------------
__device__ __forceinline__ v8f wmma32(v16h a, v16h b, v8f c) {
  // D = A(16x32 f16) * B(32x16 f16) + C(16x16 f32)
  return __builtin_amdgcn_wmma_f32_16x16x32_f16(false, a, false, b, (short)0, c,
                                                false, false);
}

// A-fragment (16x32 f16), A row-major with leading dim lda.
// Layout: lanes 0-15 row M=lane hold K 0..15; lanes 16-31 hold K 16..31.
__device__ __forceinline__ v16h load_frag_A(const _Float16* A, int lda, int m0,
                                            int k0, int lane) {
  const int r  = lane & 15;
  const int hf = lane >> 4;
  const _Float16* p = A + (size_t)(m0 + r) * lda + (k0 + hf * 16);
  v8h lo = *(const v8h*)(p);        // K = k0 + 16*hf + [0..7]
  v8h hi = *(const v8h*)(p + 8);    // K = k0 + 16*hf + 8 + [0..7]
  v16h out;
#pragma unroll
  for (int i = 0; i < 8; ++i) { out[i] = lo[i]; out[i + 8] = hi[i]; }
  return out;
}

// B-fragment (32x16 f16) from Bt = B^T stored row-major [N x K], leading dim ldb.
// Contiguous-k layout (as load_frag_A): lane n (0-15) holds col n, K=0..15; lanes 16-31 hold col n, K=16..31.
__device__ __forceinline__ v16h load_frag_B(const _Float16* Bt, int ldb, int n0,
                                            int k0, int lane) {
  const int n  = lane & 15;
  const int hf = lane >> 4;
  const _Float16* p = Bt + (size_t)(n0 + n) * ldb + (k0 + hf * 16);
  return *(const v16h*)p;           // 16 contiguous halves (32B)
}

// B-fragment from an LDS tile stored as 64 rows x 32 halves (packed, ld=32).
__device__ __forceinline__ v16h load_frag_B_lds(const _Float16* base, int n0loc,
                                                int lane) {
  const int n  = lane & 15;
  const int hf = lane >> 4;
  const _Float16* p = base + (n0loc + n) * 32 + hf * 16;
  return *(const v16h*)p;
}

// ---------------- TDM: DMA a 64(N) x 32(K) f16 tile of Bt into LDS ----------
// D# per CDNA5 ISA. 2-D tile, groups 2/3 unused (zero).
__device__ __forceinline__ void tdm_load_b_tile(_Float16* ldsDst,
                                                const _Float16* gsrc,
                                                int rowStrideElems) {
  const unsigned lds_addr = (unsigned)(unsigned long long)ldsDst;  // LDS byte off
  const unsigned long long ga = (unsigned long long)gsrc;          // global bytes
  v4u g0;
  g0[0] = 1u;                                            // count=1 (valid user D#)
  g0[1] = lds_addr;                                      // lds_addr [63:32]
  g0[2] = (unsigned)(ga & 0xFFFFFFFFu);                  // global_addr [95:64]
  g0[3] = (unsigned)((ga >> 32) & 0x01FFFFFFu) | (2u << 30);  // addr[56:32]|type=2
  v8i g1;
  g1[0] = (int)(1u << 16);        // data_size=1 (2 bytes/elem), no mask/flags
  g1[1] = (int)(32u << 16);       // tensor_dim0[15:0]=32  (bits 63:48)
  g1[2] = (int)(64u << 16);       // tensor_dim0 hi=0 | tensor_dim1[15:0]=64
  g1[3] = (int)(32u << 16);       // tensor_dim1 hi=0 | tile_dim0=32 (bits 127:112)
  g1[4] = 64;                     // tile_dim1=64, tile_dim2=0
  g1[5] = rowStrideElems;         // tensor_dim0_stride[31:0] (elements)
  g1[6] = 0;                      // stride0 hi | stride1 lo
  g1[7] = 0;                      // stride1 hi
  v4i z4 = {0, 0, 0, 0};
#if defined(__clang_major__) && __clang_major__ >= 23
  v8i z8 = {0, 0, 0, 0, 0, 0, 0, 0};
  __builtin_amdgcn_tensor_load_to_lds(g0, g1, z4, z4, z8, 0);
#else
  __builtin_amdgcn_tensor_load_to_lds(g0, g1, z4, z4, 0);
#endif
}

// ---------------- GEMM: Out[M,N] = A[M,K](f16) * Bt[N,K]^T(f16) (+epilogue) ---
enum { EPI_F16 = 0, EPI_BIAS_F32 = 1, EPI_RESID_F32 = 2, EPI_RELU_F16 = 3 };

template <int EPI>
__global__ __launch_bounds__(128) void gemm_wmma_kernel(
    const _Float16* __restrict__ A, const _Float16* __restrict__ Bt,
    const float* __restrict__ bias, const float* resid, void* outv,
    int M, int N, int K) {
  __shared__ _Float16 bTile[2][64 * 32];  // double-buffered TDM weight tile
  const int lane = threadIdx.x & 31;
  const int wave = threadIdx.x >> 5;
  const int m0 = blockIdx.y * 128 + wave * 32;  // 32 rows per wave
  const int n0 = blockIdx.x * 64;               // 64 cols per block
  v8f acc[2][4] = {};
  const int nsteps = K / 32;
  if (wave == 0) {  // prime pipeline: tile for k0 = 0
    tdm_load_b_tile(&bTile[0][0], Bt + (size_t)n0 * K, K);
  }
  for (int s = 0; s < nsteps; ++s) {
    const int k0 = s * 32;
    if (wave == 0) __builtin_amdgcn_s_wait_tensorcnt(0);  // cur tile landed
    __syncthreads();                                      // publish to all waves
    if (wave == 0 && s + 1 < nsteps) {                    // async-load next tile
      tdm_load_b_tile(&bTile[(s + 1) & 1][0], Bt + (size_t)n0 * K + (k0 + 32), K);
    }
    if (k0 + 64 < K) {  // prefetch next A line (gfx1250 global_prefetch)
      __builtin_prefetch(A + (size_t)(m0 + (lane & 15)) * K + k0 + 64, 0, 1);
    }
    v16h a0 = load_frag_A(A, K, m0, k0, lane);
    v16h a1 = load_frag_A(A, K, m0 + 16, k0, lane);
    const _Float16* bt = &bTile[s & 1][0];
#pragma unroll
    for (int c = 0; c < 4; ++c) {
      v16h bf = load_frag_B_lds(bt, 16 * c, lane);
      acc[0][c] = wmma32(a0, bf, acc[0][c]);
      acc[1][c] = wmma32(a1, bf, acc[1][c]);
    }
  }
  // C/D layout: vgpr j -> row (m0 + 16r + 8*(lane>=16) + j), col n0+16c+(lane&15)
  const int nl  = lane & 15;
  const int mb8 = (lane >> 4) * 8;
#pragma unroll
  for (int r = 0; r < 2; ++r) {
#pragma unroll
    for (int c = 0; c < 4; ++c) {
      const int n    = n0 + 16 * c + nl;
      const int mrow = m0 + 16 * r + mb8;
      float bv = 0.f;
      if (EPI == EPI_BIAS_F32 || EPI == EPI_RESID_F32 || EPI == EPI_RELU_F16)
        bv = bias[n];
#pragma unroll
      for (int j = 0; j < 8; ++j) {
        const size_t off = (size_t)(mrow + j) * N + n;
        const float a = acc[r][c][j];
        if (EPI == EPI_F16) {
          ((_Float16*)outv)[off] = (_Float16)a;
        } else if (EPI == EPI_BIAS_F32) {
          ((float*)outv)[off] = a + bv;
        } else if (EPI == EPI_RESID_F32) {
          ((float*)outv)[off] = resid[off] + a + bv;
        } else {  // EPI_RELU_F16
          const float t = a + bv;
          ((_Float16*)outv)[off] = (_Float16)(t > 0.f ? t : 0.f);
        }
      }
    }
  }
}

// ---------------- Flash attention: one wave per (b, h, 16-row tile) ----------
__global__ __launch_bounds__(32) void attn_kernel(
    const _Float16* __restrict__ q, const _Float16* __restrict__ k,
    const _Float16* __restrict__ vT, _Float16* __restrict__ att) {
  __shared__ _Float16 pLds[16 * 32];  // probability tile, row-major ld=32
  const int lane = threadIdx.x;
  const int rt = blockIdx.x, hh = blockIdx.y, bb = blockIdx.z;
  const int r0 = rt * 16;
  const _Float16* qb  = q  + (size_t)bb * Tt * Cc + hh * HSc;   // row stride Cc
  const _Float16* kb  = k  + (size_t)bb * Tt * Cc + hh * HSc;   // row stride Cc
  const _Float16* vTb = vT + (size_t)(bb * Hh + hh) * HSc * Tt; // [HS, T]

  // Q fragments for full K = HS = 64
  const v16h qa0 = load_frag_A(qb, Cc, r0, 0, lane);
  const v16h qa1 = load_frag_A(qb, Cc, r0, 32, lane);

  const int nl  = lane & 15;
  const int mb8 = (lane >> 4) * 8;
  float rmax[8], rsum[8];
  v8f o[4] = {};
#pragma unroll
  for (int j = 0; j < 8; ++j) { rmax[j] = -1e30f; rsum[j] = 0.f; }

  const float sc = 0.125f;  // 1/sqrt(64)
  for (int j0 = 0; j0 < r0 + 16; j0 += 32) {  // 32 key columns per step
    v8f zero = {};
    // scores tile0: cols j0..j0+15 ; tile1: cols j0+16..j0+31 (K = d = 64)
    v8f s0 = wmma32(qa0, load_frag_B(kb, Cc, j0, 0, lane), zero);
    s0     = wmma32(qa1, load_frag_B(kb, Cc, j0, 32, lane), s0);
    v8f s1 = zero;
    if (j0 < r0) {  // tile1 has at least one unmasked column (wave-uniform)
      s1 = wmma32(qa0, load_frag_B(kb, Cc, j0 + 16, 0, lane), zero);
      s1 = wmma32(qa1, load_frag_B(kb, Cc, j0 + 16, 32, lane), s1);
    }

    float f[8];
#pragma unroll
    for (int j = 0; j < 8; ++j) {
      const int row = r0 + mb8 + j;
      float a0v = s0[j] * sc;
      float a1v = s1[j] * sc;
      if (j0 + nl > row)      a0v = -1e30f;  // causal mask
      if (j0 + 16 + nl > row) a1v = -1e30f;
      // row max across the 16 lanes of this half-wave
      float m = fmaxf(a0v, a1v);
#pragma unroll
      for (int msk = 1; msk <= 8; msk <<= 1) m = fmaxf(m, __shfl_xor(m, msk, 32));
      const float nm = fmaxf(rmax[j], m);
      const float fj = __expf(rmax[j] - nm);
      rmax[j] = nm;
      const float e0 = __expf(a0v - nm);
      const float e1 = __expf(a1v - nm);
      float ps = e0 + e1;
#pragma unroll
      for (int msk = 1; msk <= 8; msk <<= 1) ps += __shfl_xor(ps, msk, 32);
      rsum[j] = rsum[j] * fj + ps;
      f[j] = fj;
      pLds[(mb8 + j) * 32 + nl]      = (_Float16)e0;  // C-layout -> LDS
      pLds[(mb8 + j) * 32 + 16 + nl] = (_Float16)e1;
    }
    __syncthreads();
    const v16h pf = load_frag_A(pLds, 32, 0, 0, lane);  // A-frag of P (16x32)
#pragma unroll
    for (int c = 0; c < 4; ++c) {  // rescale running O, then accumulate P@V
#pragma unroll
      for (int j = 0; j < 8; ++j) o[c][j] *= f[j];
      const v16h vf = load_frag_B(vTb, Tt, 16 * c, j0, lane);
      o[c] = wmma32(pf, vf, o[c]);
    }
    __syncthreads();
  }
#pragma unroll
  for (int c = 0; c < 4; ++c) {
#pragma unroll
    for (int j = 0; j < 8; ++j) {
      const float val = o[c][j] / rsum[j];
      att[((size_t)bb * Tt + r0 + mb8 + j) * Cc + hh * HSc + 16 * c + nl] =
          (_Float16)val;
    }
  }
}

// ---------------- LayerNorm: f32 in -> f16 out, one 256-thread block per row --
__device__ __forceinline__ float block_sum(float v, float* sbuf) {
#pragma unroll
  for (int m = 16; m > 0; m >>= 1) v += __shfl_xor(v, m, 32);
  __syncthreads();
  if ((threadIdx.x & 31) == 0) sbuf[threadIdx.x >> 5] = v;
  __syncthreads();
  float t = 0.f;
#pragma unroll
  for (int i = 0; i < 8; ++i) t += sbuf[i];
  return t;
}

__global__ __launch_bounds__(256) void ln_kernel(const float* __restrict__ x,
                                                 const float* __restrict__ g,
                                                 const float* __restrict__ bt,
                                                 _Float16* __restrict__ out) {
  __shared__ float sbuf[8];
  const int row = blockIdx.x;
  const float* xr = x + (size_t)row * Cc;
  float v[4];
  float s = 0.f;
#pragma unroll
  for (int i = 0; i < 4; ++i) { v[i] = xr[threadIdx.x + 256 * i]; s += v[i]; }
  s = block_sum(s, sbuf);
  const float mu = s * (1.0f / Cc);
  float qq = 0.f;
#pragma unroll
  for (int i = 0; i < 4; ++i) { const float d = v[i] - mu; qq += d * d; }
  qq = block_sum(qq, sbuf);
  const float rs = rsqrtf(qq * (1.0f / Cc) + 1e-5f);
#pragma unroll
  for (int i = 0; i < 4; ++i) {
    const int c = threadIdx.x + 256 * i;
    out[(size_t)row * Cc + c] = (_Float16)((v[i] - mu) * rs * g[c] + bt[c]);
  }
}

// ---------------- embedding: x = tok_emb[idx] + pos_emb ----------------------
__global__ void embed_kernel(const int* __restrict__ idx,
                             const float* __restrict__ tok,
                             const float* __restrict__ pos,
                             float* __restrict__ x) {
  const int i = blockIdx.x * blockDim.x + threadIdx.x;  // over B*T*C
  const int c  = i & (Cc - 1);
  const int bt = i >> 10;
  const int t  = bt & (Tt - 1);
  x[i] = tok[(size_t)idx[bt] * Cc + c] + pos[t * Cc + c];
}

// ---------------- weight transpose + f32 -> f16: Wt[n*K+k] = W[k*N+n] --------
__global__ void wtrans_kernel(const float* __restrict__ W,
                              _Float16* __restrict__ Wt, int K, int N) {
  const int i = blockIdx.x * blockDim.x + threadIdx.x;  // over N*K
  const int kk = i % K;
  const int n  = i / K;
  Wt[i] = (_Float16)W[(size_t)kk * N + n];
}

// ---------------- v transpose: vT[((b*H+h)*HS+d)*T + t] = v[b,t,h,d] ---------
__global__ void vtrans_kernel(const _Float16* __restrict__ v,
                              _Float16* __restrict__ vT) {
  const int i = blockIdx.x * blockDim.x + threadIdx.x;  // over B*H*HS*T
  const int t  = i & (Tt - 1);
  const int d  = (i >> 10) & (HSc - 1);
  const int bh = i >> 16;  // HS*T = 65536
  const int b  = bh >> 4;
  const int h  = bh & (Hh - 1);
  vT[i] = v[((size_t)(b * Tt + t)) * Cc + h * HSc + d];
}

// ---------------- host orchestration -----------------------------------------
extern "C" void kernel_launch(void* const* d_in, const int* in_sizes, int n_in,
                              void* d_out, int out_size, void* d_ws,
                              size_t ws_size, hipStream_t stream) {
  (void)in_sizes; (void)n_in; (void)out_size; (void)ws_size;
  const int*   idx     = (const int*)  d_in[0];
  const float* tok_emb = (const float*)d_in[1];
  const float* pos_emb = (const float*)d_in[2];
  const float* Wq      = (const float*)d_in[3];
  const float* Wk      = (const float*)d_in[4];
  const float* Wv      = (const float*)d_in[5];
  const float* Wo      = (const float*)d_in[6];
  const float* bo      = (const float*)d_in[7];
  const float* ln1_g   = (const float*)d_in[8];
  const float* ln1_b   = (const float*)d_in[9];
  const float* ln2_g   = (const float*)d_in[10];
  const float* ln2_b   = (const float*)d_in[11];
  const float* W1      = (const float*)d_in[12];
  const float* b1      = (const float*)d_in[13];
  const float* W2      = (const float*)d_in[14];
  const float* b2      = (const float*)d_in[15];
  const float* lnf_g   = (const float*)d_in[16];
  const float* lnf_b   = (const float*)d_in[17];
  const float* Whead   = (const float*)d_in[18];
  const float* bhead   = (const float*)d_in[19];

  char* ws = (char*)d_ws;
  size_t off = 0;
  float*    x    = (float*)(ws + off);    off += (size_t)Mm * Cc * 4;  // 32 MB
  _Float16* hbuf = (_Float16*)(ws + off); off += (size_t)Mm * Cc * 2;  // 16 MB
  _Float16* qbuf = (_Float16*)(ws + off); off += (size_t)Mm * Cc * 2;
  _Float16* kbuf = (_Float16*)(ws + off); off += (size_t)Mm * Cc * 2;
  _Float16* vbuf = (_Float16*)(ws + off); off += (size_t)Mm * Cc * 2;
  _Float16* vTb  = (_Float16*)(ws + off); off += (size_t)Mm * Cc * 2;
  _Float16* Wt   = (_Float16*)(ws + off); off += (size_t)4 * Cc * Cc * 2; // 8 MB
  _Float16* ubuf = qbuf;  // MLP mid [8192 x 4096] aliases q/k/v/vT (64 MB)
  _Float16* attb = hbuf;  // attention out aliases h

  const dim3 gblk(128);
  const dim3 gC (Cc / 64,     Mm / 128);  // N=1024 GEMMs
  const dim3 g4C(4 * Cc / 64, Mm / 128);  // N=4096 GEMM
  const dim3 gV (Vv / 64,     Mm / 128);  // head GEMM

  embed_kernel<<<(Mm * Cc) / 256, 256, 0, stream>>>(idx, tok_emb, pos_emb, x);

  for (int l = 0; l < Ll; ++l) {
    const size_t wl  = (size_t)l * Cc * Cc;
    const size_t wl4 = (size_t)l * Cc * 4 * Cc;
    ln_kernel<<<Mm, 256, 0, stream>>>(x, ln1_g + l * Cc, ln1_b + l * Cc, hbuf);

    wtrans_kernel<<<(Cc * Cc) / 256, 256, 0, stream>>>(Wq + wl, Wt, Cc, Cc);
    gemm_wmma_kernel<EPI_F16><<<gC, gblk, 0, stream>>>(hbuf, Wt, nullptr, nullptr,
                                                       qbuf, Mm, Cc, Cc);
    wtrans_kernel<<<(Cc * Cc) / 256, 256, 0, stream>>>(Wk + wl, Wt, Cc, Cc);
    gemm_wmma_kernel<EPI_F16><<<gC, gblk, 0, stream>>>(hbuf, Wt, nullptr, nullptr,
                                                       kbuf, Mm, Cc, Cc);
    wtrans_kernel<<<(Cc * Cc) / 256, 256, 0, stream>>>(Wv + wl, Wt, Cc, Cc);
    gemm_wmma_kernel<EPI_F16><<<gC, gblk, 0, stream>>>(hbuf, Wt, nullptr, nullptr,
                                                       vbuf, Mm, Cc, Cc);

    vtrans_kernel<<<(Mm * Cc) / 256, 256, 0, stream>>>(vbuf, vTb);
    attn_kernel<<<dim3(Tt / 16, Hh, Bb), 32, 0, stream>>>(qbuf, kbuf, vTb, attb);

    wtrans_kernel<<<(Cc * Cc) / 256, 256, 0, stream>>>(Wo + wl, Wt, Cc, Cc);
    gemm_wmma_kernel<EPI_RESID_F32><<<gC, gblk, 0, stream>>>(
        attb, Wt, bo + l * Cc, x, x, Mm, Cc, Cc);

    ln_kernel<<<Mm, 256, 0, stream>>>(x, ln2_g + l * Cc, ln2_b + l * Cc, hbuf);

    wtrans_kernel<<<(Cc * 4 * Cc) / 256, 256, 0, stream>>>(W1 + wl4, Wt, Cc, 4 * Cc);
    gemm_wmma_kernel<EPI_RELU_F16><<<g4C, gblk, 0, stream>>>(
        hbuf, Wt, b1 + (size_t)l * 4 * Cc, nullptr, ubuf, Mm, 4 * Cc, Cc);

    wtrans_kernel<<<(Cc * 4 * Cc) / 256, 256, 0, stream>>>(W2 + wl4, Wt, 4 * Cc, Cc);
    gemm_wmma_kernel<EPI_RESID_F32><<<gC, gblk, 0, stream>>>(
        ubuf, Wt, b2 + l * Cc, x, x, Mm, Cc, 4 * Cc);
  }

  ln_kernel<<<Mm, 256, 0, stream>>>(x, lnf_g, lnf_b, hbuf);
  wtrans_kernel<<<(Cc * Vv) / 256, 256, 0, stream>>>(Whead, Wt, Cc, Vv);
  gemm_wmma_kernel<EPI_BIAS_F32><<<gV, gblk, 0, stream>>>(
      hbuf, Wt, bhead, nullptr, d_out, Mm, Vv, Cc);
}
